// SwitchTransformerEncoderLayer_80814104642415
// MI455X (gfx1250) — matrix operand order repaired and checked by reading, not compiled, not run on hardware
//
#include <hip/hip_runtime.h>

// ---------------------------------------------------------------- constants
constexpr int cL  = 1024;
constexpr int cN  = 16;
constexpr int cD  = 1024;
constexpr int cH  = 4;
constexpr int cHD = 256;
constexpr int cE  = 8;
constexpr int cFF = 4096;
constexpr int cS  = cL * cN;          // 16384 tokens
constexpr int cCAP = cS / cE;         // 2048 capacity per expert

typedef __attribute__((ext_vector_type(16))) __bf16 v16bf;
typedef __attribute__((ext_vector_type(8)))  float  v8f;

#if defined(__AMDGCN__) && __has_builtin(__builtin_amdgcn_global_load_async_to_lds_b128)
#define HAVE_ASYNC_LDS 1
#endif

// ------------------------------------------------------------- bf16 helpers
static __device__ __forceinline__ unsigned short f2us(float f) {
  union { float f; unsigned u; } v; v.f = f;
  unsigned r = v.u + 0x7FFFu + ((v.u >> 16) & 1u);   // round-to-nearest-even
  return (unsigned short)(r >> 16);
}
static __device__ __forceinline__ __bf16 us2bf(unsigned short s) {
  union { unsigned short u; __bf16 b; } v; v.u = s; return v.b;
}
static __device__ __forceinline__ unsigned short to_us(float v) { return f2us(v); }
static __device__ __forceinline__ unsigned short to_us(unsigned short v) { return v; }
static __device__ __forceinline__ void store_out(float* p, float v) { *p = v; }
static __device__ __forceinline__ void store_out(unsigned short* p, float v) { *p = f2us(v); }

#if defined(HAVE_ASYNC_LDS)
typedef int v4i_async __attribute__((vector_size(16)));
static __device__ __forceinline__ void async_copy_b128(const void* g, void* l) {
  __builtin_amdgcn_global_load_async_to_lds_b128(
      (__attribute__((address_space(1))) v4i_async*)g,
      (__attribute__((address_space(3))) v4i_async*)l, 0, 0);
}
static __device__ __forceinline__ void wait_async() {
#if __has_builtin(__builtin_amdgcn_s_wait_asynccnt)
  __builtin_amdgcn_s_wait_asynccnt(0);
#else
  asm volatile("s_wait_asynccnt 0" ::: "memory");
#endif
}
#endif

// --------------------------------------------------------------- WMMA core
static __device__ __forceinline__ v8f wmma_bf16(v16bf a, v16bf b, v8f c) {
  return __builtin_amdgcn_wmma_f32_16x16x32_bf16(
      /*neg_a=*/false, a, /*neg_b=*/false, b,
      /*c_mod=*/(short)0, c, /*reuse_a=*/false, /*reuse_b=*/false);
}

// A fragment 16x32 (MxK), bf16 in LDS (row-major, stride lda elements)
static __device__ __forceinline__ v16bf frag_a(const unsigned short* p, int lda, int lane) {
  const int half = lane >> 4, row = lane & 15;
  const unsigned short* r = p + row * lda;
  v16bf a;
#pragma unroll
  for (int v = 0; v < 8; ++v) {
    int k0 = (v < 4) ? (half * 8 + 2 * v) : (16 + half * 8 + 2 * (v - 4));
    a[2 * v]     = us2bf(r[k0]);
    a[2 * v + 1] = us2bf(r[k0 + 1]);
  }
  return a;
}
// B fragment 32x16 (KxN), bf16 in LDS stored [K][N] with stride ldb elements
static __device__ __forceinline__ v16bf frag_b(const unsigned short* p, int ldb, int lane) {
  const int half = lane >> 4;
  const unsigned short* r = p + (lane & 15);
  v16bf b;
#pragma unroll
  for (int v = 0; v < 8; ++v) {
    int k0 = (v < 4) ? (half * 8 + 2 * v) : (16 + half * 8 + 2 * (v - 4));
    b[2 * v]     = us2bf(r[k0 * ldb]);
    b[2 * v + 1] = us2bf(r[(k0 + 1) * ldb]);
  }
  return b;
}

// -------------------------------------------------------- generic WMMA GEMM
// C[M,N] = A[M,K] * op(B) + bias ; TRANSB==1 -> B is [N,K] (use B^T), else [K,N]
// grid.z batches (experts) via strides.
constexpr int BM = 128, BN = 128, BK = 64;

template <typename AT, int TRANSB, typename CT>
__global__ __launch_bounds__(256)
void gemm_wmma(const AT* __restrict__ A, const float* __restrict__ B,
               const float* __restrict__ bias, CT* __restrict__ C,
               int M, int N, int K,
               long long sA, long long sB, long long sBias, long long sC) {
  __shared__ __attribute__((aligned(16))) unsigned short As[BM][BK + 8]; // 18 KB
  __shared__ __attribute__((aligned(16))) unsigned short Bs[BK][BN + 8]; // 17.4 KB

  const int e = blockIdx.z;
  A += (long long)e * sA;
  B += (long long)e * sB;
  C += (long long)e * sC;
  const float* bp = bias ? (bias + (long long)e * sBias) : nullptr;

  const int n0 = blockIdx.x * BN, m0 = blockIdx.y * BM;
  const int t = threadIdx.x, lane = t & 31, w = t >> 5;
  const int wm = (w & 3) * 32, wn = (w >> 2) * 64;

  v8f acc[2][4] = {};

  for (int k0 = 0; k0 < K; k0 += BK) {
    // ---- stage A tile (128 x 64 bf16)
    if constexpr (sizeof(AT) == 2) {
#if defined(HAVE_ASYNC_LDS)
      // pure bf16 copy: async global->LDS, 16B per lane per issue
#pragma unroll
      for (int it = 0; it < (BM * BK) / (256 * 8); ++it) {   // 4 issues
        int fi = it * 256 + t;
        int r = fi >> 3, cc = (fi & 7) * 8;
        async_copy_b128((const void*)&A[(long long)(m0 + r) * K + k0 + cc],
                        (void*)&As[r][cc]);
      }
#else
      for (int i = t; i < BM * BK; i += 256) {
        int r = i >> 6, c = i & 63;
        As[r][c] = to_us(A[(long long)(m0 + r) * K + k0 + c]);
      }
#endif
    } else {
      for (int i = t; i < BM * BK; i += 256) {
        int r = i >> 6, c = i & 63;
        As[r][c] = to_us(A[(long long)(m0 + r) * K + k0 + c]);
      }
    }
    // ---- stage B tile (64 x 128, fp32 -> bf16)
    if (TRANSB) {
      for (int i = t; i < BK * BN; i += 256) {
        int n = i >> 6, k = i & 63;
        Bs[k][n] = f2us(B[(long long)(n0 + n) * K + k0 + k]);
      }
    } else {
      for (int i = t; i < BK * BN; i += 256) {
        int k = i >> 7, n = i & 127;
        Bs[k][n] = f2us(B[(long long)(k0 + k) * N + n0 + n]);
      }
    }
    if (k0 + BK < K) {   // hint next A tile into cache (global_prefetch_b8)
      __builtin_prefetch(&A[(long long)(m0 + (t >> 1)) * K + k0 + BK + (t & 1) * 32], 0, 0);
    }
#if defined(HAVE_ASYNC_LDS)
    if constexpr (sizeof(AT) == 2) wait_async();
#endif
    __syncthreads();

#pragma unroll
    for (int kk = 0; kk < BK; kk += 32) {
      v16bf af[2], bfr[4];
#pragma unroll
      for (int mt = 0; mt < 2; ++mt) af[mt] = frag_a(&As[wm + mt * 16][kk], BK + 8, lane);
#pragma unroll
      for (int nt = 0; nt < 4; ++nt) bfr[nt] = frag_b(&Bs[kk][wn + nt * 16], BN + 8, lane);
#pragma unroll
      for (int mt = 0; mt < 2; ++mt)
#pragma unroll
        for (int nt = 0; nt < 4; ++nt)
          acc[mt][nt] = wmma_bf16(af[mt], bfr[nt], acc[mt][nt]);
    }
    __syncthreads();
  }

  const int half = lane >> 4, nl = lane & 15;
#pragma unroll
  for (int mt = 0; mt < 2; ++mt)
#pragma unroll
    for (int nt = 0; nt < 4; ++nt) {
      int col = n0 + wn + nt * 16 + nl;
      float bv = bp ? bp[col] : 0.f;
#pragma unroll
      for (int v = 0; v < 8; ++v) {
        int row = m0 + wm + mt * 16 + v + 8 * half;
        store_out(&C[(long long)row * N + col], acc[mt][nt][v] + bv);
      }
    }
}

// --------------------------------------------------- flash attention kernel
// grid: (L/64, N*H); block 128 (4 waves). Each wave owns 16 query rows.
__global__ __launch_bounds__(128)
void attn_kernel(const unsigned short* __restrict__ qkv, unsigned short* __restrict__ out) {
  const int nh = blockIdx.y, n = nh >> 2, h = nh & 3;
  const int q0 = blockIdx.x * 64;
  const int t = threadIdx.x, lane = t & 31, w = t >> 5;
  const int qc = h * cHD, kc = cD + h * cHD, vc = 2 * cD + h * cHD;

  __shared__ unsigned short Ks[cHD][32 + 2];   // K^T tile [d][key]   ~17 KB
  __shared__ unsigned short Vs[32][cHD + 8];   // V tile  [key][d]    ~16.5 KB
  __shared__ float          Ssc[4][16][32 + 2];
  __shared__ unsigned short Pb[4][16][32];
  __shared__ float rowm[4][16], rows[4][16], rowf[4][16];

  // resident Q fragments: wave w -> rows q0 + w*16 .. +15, 8 K-chunks of 32
  const int qrow0 = q0 + w * 16;
  v16bf qf[8];
  {
    const int half = lane >> 4;
    const int row = qrow0 + (lane & 15);
    const long long base = ((long long)row * cN + n) * (3 * cD) + qc;
#pragma unroll
    for (int kk = 0; kk < 8; ++kk) {
      v16bf a;
#pragma unroll
      for (int v = 0; v < 8; ++v) {
        int kofs = kk * 32 + ((v < 4) ? (half * 8 + 2 * v) : (16 + half * 8 + 2 * (v - 4)));
        a[2 * v]     = us2bf(qkv[base + kofs]);
        a[2 * v + 1] = us2bf(qkv[base + kofs + 1]);
      }
      qf[kk] = a;
    }
  }

  v8f O[16] = {};
  if (lane < 16) { rowm[w][lane] = -1e30f; rows[w][lane] = 0.f; }
  const float scale = 0.0625f;  // 1/sqrt(256)

  for (int kb = 0; kb < cL; kb += 32) {
    __syncthreads();                               // protect K/V reuse
    for (int i = t; i < 32 * cHD; i += 128) {      // K^T stage (transpose)
      int j = i >> 8, d = i & 255;
      long long s = (long long)(kb + j) * cN + n;
      Ks[d][j] = qkv[s * (3 * cD) + kc + d];
    }
    for (int i = t; i < 32 * cHD; i += 128) {      // V stage
      int j = i >> 8, d = i & 255;
      long long s = (long long)(kb + j) * cN + n;
      Vs[j][d] = qkv[s * (3 * cD) + vc + d];
    }
    __syncthreads();

    // S = Q * K^T  (16 x 32 per wave)
    v8f sacc[2] = {};
#pragma unroll
    for (int kk = 0; kk < 8; ++kk) {
      sacc[0] = wmma_bf16(qf[kk], frag_b(&Ks[kk * 32][0], 34, lane), sacc[0]);
      sacc[1] = wmma_bf16(qf[kk], frag_b(&Ks[kk * 32][16], 34, lane), sacc[1]);
    }
    const int half = lane >> 4, nl = lane & 15;
#pragma unroll
    for (int v = 0; v < 8; ++v) {
      Ssc[w][v + 8 * half][nl]      = sacc[0][v] * scale;
      Ssc[w][v + 8 * half][16 + nl] = sacc[1][v] * scale;
    }

    // online softmax (16 lanes, one per query row)
    if (lane < 16) {
      int r = lane;
      float mold = rowm[w][r], mnew = mold;
#pragma unroll
      for (int j = 0; j < 32; ++j) mnew = fmaxf(mnew, Ssc[w][r][j]);
      float fac = __expf(mold - mnew), sum = 0.f;
#pragma unroll
      for (int j = 0; j < 32; ++j) {
        float p = __expf(Ssc[w][r][j] - mnew);
        Pb[w][r][j] = f2us(p);
        sum += p;
      }
      rowm[w][r] = mnew;
      rows[w][r] = rows[w][r] * fac + sum;
      rowf[w][r] = fac;
    }

    // rescale O, then O += P * V
    float f0[8];
#pragma unroll
    for (int v = 0; v < 8; ++v) f0[v] = rowf[w][v + 8 * half];
#pragma unroll
    for (int td = 0; td < 16; ++td)
#pragma unroll
      for (int v = 0; v < 8; ++v) O[td][v] *= f0[v];

    v16bf pf = frag_a(&Pb[w][0][0], 32, lane);
#pragma unroll
    for (int td = 0; td < 16; ++td)
      O[td] = wmma_bf16(pf, frag_b(&Vs[0][td * 16], cHD + 8, lane), O[td]);
  }

  // finalize: O / rowsum -> global (bf16)
  {
    const int half = lane >> 4, nl = lane & 15;
    float inv[8];
#pragma unroll
    for (int v = 0; v < 8; ++v) inv[v] = 1.f / rows[w][v + 8 * half];
#pragma unroll
    for (int td = 0; td < 16; ++td)
#pragma unroll
      for (int v = 0; v < 8; ++v) {
        int row = qrow0 + v + 8 * half;
        long long s = (long long)row * cN + n;
        out[s * cD + h * cHD + td * 16 + nl] = f2us(O[td][v] * inv[v]);
      }
  }
}

// --------------------------------------------------- residual + LayerNorm
__global__ __launch_bounds__(256)
void add_ln_kernel(const float* __restrict__ x, const float* __restrict__ dlt,
                   const float* __restrict__ g, const float* __restrict__ b,
                   float* __restrict__ out) {
  const int token = blockIdx.x * 8 + (threadIdx.x >> 5);
  const int lane = threadIdx.x & 31;
  const long long base = (long long)token * cD;
  float vals[32], sum = 0.f;
#pragma unroll
  for (int i = 0; i < 32; ++i) {
    int c = lane + i * 32;
    float v = x[base + c] + dlt[base + c];
    vals[i] = v; sum += v;
  }
  for (int o = 16; o > 0; o >>= 1) sum += __shfl_xor(sum, o, 32);
  float mu = sum * (1.f / cD), var = 0.f;
#pragma unroll
  for (int i = 0; i < 32; ++i) { float d = vals[i] - mu; var += d * d; }
  for (int o = 16; o > 0; o >>= 1) var += __shfl_xor(var, o, 32);
  float r = rsqrtf(var * (1.f / cD) + 1e-5f);
#pragma unroll
  for (int i = 0; i < 32; ++i) {
    int c = lane + i * 32;
    out[base + c] = (vals[i] - mu) * r * g[c] + b[c];
  }
}

// ------------------------------------------------------------- MoE gating
__global__ __launch_bounds__(256)
void gate_kernel(const float* __restrict__ y1, const float* __restrict__ wg,
                 int* __restrict__ idx, float* __restrict__ gateval) {
  const int token = blockIdx.x * 8 + (threadIdx.x >> 5);
  const int lane = threadIdx.x & 31;
  const long long base = (long long)token * cD;
  float lg[cE];
#pragma unroll
  for (int e = 0; e < cE; ++e) lg[e] = 0.f;
  for (int d = lane; d < cD; d += 32) {
    float xv = y1[base + d];
#pragma unroll
    for (int e = 0; e < cE; ++e) lg[e] += xv * wg[d * cE + e];
  }
#pragma unroll
  for (int e = 0; e < cE; ++e)
    for (int o = 16; o > 0; o >>= 1) lg[e] += __shfl_xor(lg[e], o, 32);
  if (lane == 0) {
    float mx = lg[0]; int bi = 0;
#pragma unroll
    for (int e = 1; e < cE; ++e) if (lg[e] > mx) { mx = lg[e]; bi = e; }
    float den = 0.f;
#pragma unroll
    for (int e = 0; e < cE; ++e) den += __expf(lg[e] - mx);
    idx[token] = bi;
    gateval[token] = 1.f / den;   // softmax value at argmax
  }
}

// token-order prefix position per expert -> slot (E*CAP sentinel if dropped)
__global__ __launch_bounds__(256)
void scan_kernel(const int* __restrict__ idx, int* __restrict__ slot) {
  const int e = threadIdx.x >> 5;       // 8 waves, one expert each
  const int lane = threadIdx.x & 31;
  int count = 0;
  for (int s0 = 0; s0 < cS; s0 += 32) {
    int s = s0 + lane;
    bool mine = (idx[s] == e);
    unsigned mask = __builtin_amdgcn_ballot_w32(mine);
    int loc = count + __popc(mask & ((1u << lane) - 1u));
    if (mine) slot[s] = (loc < cCAP) ? (e * cCAP + loc) : (cE * cCAP);
    count += __popc(mask);
  }
}

__global__ __launch_bounds__(256)
void dispatch_kernel(const float* __restrict__ y1, const int* __restrict__ slot,
                     unsigned short* __restrict__ disp) {
  const int token = blockIdx.x * 8 + (threadIdx.x >> 5);
  const int lane = threadIdx.x & 31;
  const int sl = slot[token];
  if (sl >= cE * cCAP) return;
  const long long src = (long long)token * cD, dst = (long long)sl * cD;
  for (int i = lane; i < cD; i += 32) disp[dst + i] = f2us(y1[src + i]);
}

// --------------------------------------------- combine + residual + LN2
__global__ __launch_bounds__(256)
void combine_ln_kernel(const float* __restrict__ y1, const float* __restrict__ yexp,
                       const int* __restrict__ slot, const float* __restrict__ gate,
                       const float* __restrict__ g, const float* __restrict__ b,
                       float* __restrict__ out) {
  const int token = blockIdx.x * 8 + (threadIdx.x >> 5);
  const int lane = threadIdx.x & 31;
  const long long base = (long long)token * cD;
  const int sl = slot[token];
  const bool kept = sl < cE * cCAP;
  const float gv = kept ? gate[token] : 0.f;
  const float* yrow = yexp + (long long)(kept ? sl : 0) * cD;

  float vals[32], sum = 0.f;
#pragma unroll
  for (int i = 0; i < 32; ++i) {
    int c = lane + i * 32;
    float v = y1[base + c] + (kept ? gv * yrow[c] : 0.f);
    vals[i] = v; sum += v;
  }
  for (int o = 16; o > 0; o >>= 1) sum += __shfl_xor(sum, o, 32);
  float mu = sum * (1.f / cD), var = 0.f;
#pragma unroll
  for (int i = 0; i < 32; ++i) { float d = vals[i] - mu; var += d * d; }
  for (int o = 16; o > 0; o >>= 1) var += __shfl_xor(var, o, 32);
  float r = rsqrtf(var * (1.f / cD) + 1e-5f);
#pragma unroll
  for (int i = 0; i < 32; ++i) {
    int c = lane + i * 32;
    out[base + c] = (vals[i] - mu) * r * g[c] + b[c];
  }
}

// ------------------------------------------------------------------- host
extern "C" void kernel_launch(void* const* d_in, const int* in_sizes, int n_in,
                              void* d_out, int out_size, void* d_ws, size_t ws_size,
                              hipStream_t stream) {
  const float* x      = (const float*)d_in[0];
  const float* ipw    = (const float*)d_in[1];
  const float* ipb    = (const float*)d_in[2];
  const float* opw    = (const float*)d_in[3];
  const float* opb    = (const float*)d_in[4];
  const float* ln1g   = (const float*)d_in[5];
  const float* ln1b   = (const float*)d_in[6];
  const float* ln2g   = (const float*)d_in[7];
  const float* ln2b   = (const float*)d_in[8];
  const float* wg     = (const float*)d_in[9];
  const float* w1     = (const float*)d_in[10];
  const float* b1     = (const float*)d_in[11];
  const float* w2     = (const float*)d_in[12];
  const float* b2     = (const float*)d_in[13];

  char* ws = (char*)d_ws;
  // region A (0..128MB): qkv bf16 -> proj f32 -> h bf16 (sequentially aliased)
  unsigned short* qkv  = (unsigned short*)ws;
  float*          proj = (float*)ws;
  unsigned short* hbuf = (unsigned short*)ws;
  // region B (128..192MB): attn bf16 -> yexp f32
  const size_t offB = 128ull << 20;
  unsigned short* attn = (unsigned short*)(ws + offB);
  float*          yexp = (float*)(ws + offB);
  // region C (192..256MB): y1 f32
  const size_t offC = 192ull << 20;
  float* y1 = (float*)(ws + offC);
  // region D (256..288MB): disp bf16
  const size_t offD = 256ull << 20;
  unsigned short* disp = (unsigned short*)(ws + offD);
  // region E: gating scalars
  const size_t offE = 288ull << 20;
  int*   idx  = (int*)(ws + offE);
  float* gate = (float*)(ws + offE + 65536);
  int*   slot = (int*)(ws + offE + 2 * 65536);

  const dim3 blk(256);

  // 1) QKV projection: [S,3D] = x[S,D] @ ipw^T + ipb  (bf16 out)
  gemm_wmma<float, 1, unsigned short>
      <<<dim3((3 * cD) / BN, cS / BM, 1), blk, 0, stream>>>(
          x, ipw, ipb, qkv, cS, 3 * cD, cD, 0, 0, 0, 0);

  // 2) flash attention
  attn_kernel<<<dim3(cL / 64, cN * cH), dim3(128), 0, stream>>>(qkv, attn);

  // 3) output projection (f32 out)
  gemm_wmma<unsigned short, 1, float>
      <<<dim3(cD / BN, cS / BM, 1), blk, 0, stream>>>(
          attn, opw, opb, proj, cS, cD, cD, 0, 0, 0, 0);

  // 4) y1 = LN1(x + proj)
  add_ln_kernel<<<cS / 8, blk, 0, stream>>>(x, proj, ln1g, ln1b, y1);

  // 5) gating + deterministic capacity scan + dispatch (bf16)
  gate_kernel<<<cS / 8, blk, 0, stream>>>(y1, wg, idx, gate);
  scan_kernel<<<1, blk, 0, stream>>>(idx, slot);
  dispatch_kernel<<<cS / 8, blk, 0, stream>>>(y1, slot, disp);

  // 6) expert grouped GEMMs: h = disp@w1 + b1 ; yexp = h@w2 + b2
  gemm_wmma<unsigned short, 0, unsigned short>
      <<<dim3(cFF / BN, cCAP / BM, cE), blk, 0, stream>>>(
          disp, w1, b1, hbuf, cCAP, cFF, cD,
          (long long)cCAP * cD, (long long)cD * cFF, cFF, (long long)cCAP * cFF);
  gemm_wmma<unsigned short, 0, float>
      <<<dim3(cD / BN, cCAP / BM, cE), blk, 0, stream>>>(
          hbuf, w2, b2, yexp, cCAP, cD, cFF,
          (long long)cCAP * cFF, (long long)cFF * cD, cD, (long long)cCAP * cD);

  // 7) out = LN2(y1 + combine)
  combine_ln_kernel<<<cS / 8, blk, 0, stream>>>(y1, yexp, slot, gate, ln2g, ln2b,
                                                (float*)d_out);
}
